// RNN_CONCRETE_37005438222866
// MI455X (gfx1250) — compile-verified, matrix operand order checked by reading, not run on hardware
//
#include <hip/hip_runtime.h>
#include <math.h>

// ---------------------------------------------------------------------------
// Types for CDNA5 WMMA / TDM (wave32, gfx1250)
// ---------------------------------------------------------------------------
typedef __bf16 bf16;
typedef bf16  v16bf __attribute__((ext_vector_type(16)));
typedef float v8f   __attribute__((ext_vector_type(8)));
typedef unsigned int u32x4 __attribute__((ext_vector_type(4)));
typedef int          i32x8 __attribute__((ext_vector_type(8)));
typedef int          i32x4 __attribute__((ext_vector_type(4)));

#define B_SZ   64
#define T_SZ   512
#define BT     (B_SZ * T_SZ)        // 32768
#define DDIM   129
#define NH     100
#define NS     100
#define DETD   5

// K-index map for the 16-bit A/B 16x32 WMMA operand layout (ISA 7.12.2):
// lane group g (=lane>>4): VGPR pair p in 0..3 -> K = 8g + 2p + w
//                          VGPR pair p in 4..7 -> K = 16 + 8g + 2(p-4) + w
__device__ __forceinline__ int kmap(int e, int g) {
    int p = e >> 1, w = e & 1;
    return ((p >> 2) << 4) + (g << 3) + ((p & 3) << 1) + w;
}

// ---------------------------------------------------------------------------
// Tiled WMMA GEMM:  out = act( X(MxKp,bf16) @ W(NpxKp,bf16)^T + bias )
// W is row-padded to Np (multiple of 64) with zero rows -> branch-free staging.
// mode 0: fp32 output (ld = N), no activation
// mode 1: bf16 output padded to ldOutB (cols >= N are 0 after ReLU)
// Block: 256 threads (8 waves, 4x2). Block tile 128x64, wave tile 32x32
// (2x2 of 16x16 WMMA tiles), K staged 32 at a time via the Tensor Data Mover.
// ---------------------------------------------------------------------------
#define GBM 128
#define GBN 64
#define GBK 32

#if defined(__AMDGCN__) && __has_builtin(__builtin_amdgcn_tensor_load_to_lds)
#define USE_TDM 1
#else
#define USE_TDM 0
#endif

__global__ __launch_bounds__(256) void wmma_gemm(
    const bf16* __restrict__ X, const bf16* __restrict__ W,
    const float* __restrict__ bias,
    float* __restrict__ outF, bf16* __restrict__ outB,
    int M, int N, int Np, int Kp, int ldX, int ldW, int ldOutB, int mode)
{
    __shared__ __align__(16) bf16 Asm[GBM * GBK];   // 8 KB
    __shared__ __align__(16) bf16 Bsm[GBN * GBK];   // 4 KB

    const int tid  = threadIdx.x;
    const int lane = tid & 31;
    const int wave = tid >> 5;   // 0..7
    const int wm   = wave & 3;   // row wave (32 rows each)
    const int wn   = wave >> 2;  // col wave (32 cols each)
    const int g    = lane >> 4;  // lane group
    const int li   = lane & 15;

    const int rowBase = blockIdx.x * GBM;
    const int colBase = blockIdx.y * GBN;

    const v8f vzero = {0.f, 0.f, 0.f, 0.f, 0.f, 0.f, 0.f, 0.f};
    v8f acc[2][2] = {{vzero, vzero}, {vzero, vzero}};

#if USE_TDM
    const unsigned int ldsA = (unsigned int)(size_t)(void*)Asm;  // LDS offset (flat low 32b)
    const unsigned int ldsB = (unsigned int)(size_t)(void*)Bsm;
#endif

    for (int k0 = 0; k0 < Kp; k0 += GBK) {
#if USE_TDM
        // ---- TDM staging: wave 0 issues two tensor DMA descriptors -------
        if (wave == 0) {
            const i32x4 zz = {0, 0, 0, 0};
            const i32x8 z8 = {0, 0, 0, 0, 0, 0, 0, 0};
            // A tile: 128 rows x 32 cols of bf16, row stride ldX
            {
                unsigned long long ga = (unsigned long long)(size_t)X
                    + 2ull * ((unsigned long long)rowBase * (unsigned)ldX + (unsigned)k0);
                u32x4 g0;
                g0[0] = 1u;                                  // count = 1 valid D#
                g0[1] = ldsA;                                // lds_addr
                g0[2] = (unsigned int)ga;                    // global_addr lo
                g0[3] = (unsigned int)(ga >> 32) | (2u << 30); // global_addr hi | type=2
                i32x8 g1;
                g1[0] = 0x00010000;                          // data_size = 1 (2 bytes)
                g1[1] = (int)(((unsigned)ldX & 0xFFFFu) << 16);        // tensor_dim0 lo
                g1[2] = (int)(((unsigned)ldX >> 16) | ((unsigned)(M & 0xFFFF) << 16));
                g1[3] = (int)(((unsigned)M >> 16) | ((unsigned)GBK << 16)); // dim1 hi | tile_dim0
                g1[4] = GBM;                                 // tile_dim1 = 128, tile_dim2 = 0
                g1[5] = ldX;                                 // tensor_dim0_stride lo32
                g1[6] = 0;                                   // stride0 hi16 | stride1 lo16
                g1[7] = 0;                                   // stride1 hi
                __builtin_amdgcn_tensor_load_to_lds(g0, g1, zz, zz, z8, 0);
            }
            // B tile: 64 rows x 32 cols of bf16, row stride ldW
            {
                unsigned long long gb = (unsigned long long)(size_t)W
                    + 2ull * ((unsigned long long)colBase * (unsigned)ldW + (unsigned)k0);
                u32x4 g0;
                g0[0] = 1u;
                g0[1] = ldsB;
                g0[2] = (unsigned int)gb;
                g0[3] = (unsigned int)(gb >> 32) | (2u << 30);
                i32x8 g1;
                g1[0] = 0x00010000;
                g1[1] = (int)(((unsigned)ldW & 0xFFFFu) << 16);
                g1[2] = (int)(((unsigned)ldW >> 16) | ((unsigned)(Np & 0xFFFF) << 16));
                g1[3] = (int)(((unsigned)Np >> 16) | ((unsigned)GBK << 16));
                g1[4] = GBN;                                 // tile_dim1 = 64
                g1[5] = ldW;
                g1[6] = 0;
                g1[7] = 0;
                __builtin_amdgcn_tensor_load_to_lds(g0, g1, zz, zz, z8, 0);
            }
            __builtin_amdgcn_s_wait_tensorcnt(0);
        }
        __syncthreads();
#else
        // ---- fallback staging: vectorized global->LDS copies -------------
        {
            int r = tid >> 1;                 // 0..127
            int c = (tid & 1) << 4;           // 0 or 16
            const bf16* src = X + (size_t)(rowBase + r) * ldX + k0 + c;
            *(uint4*)&Asm[r * GBK + c] = *(const uint4*)src;
            if (k0 + GBK < Kp) __builtin_prefetch(src + GBK, 0, 1);
        }
        {
            int r = tid >> 2;                 // 0..63
            int c = (tid & 3) << 3;           // 0,8,16,24
            const bf16* src = W + (size_t)(colBase + r) * ldW + k0 + c;
            *(uint4*)&Bsm[r * GBK + c] = *(const uint4*)src;
        }
        __syncthreads();
#endif
        // ---- 2x2 register-blocked WMMA ------------------------------------
        v16bf a0, a1, b0, b1;
        #pragma unroll
        for (int e = 0; e < 16; ++e) {
            int k = kmap(e, g);
            a0[e] = Asm[(wm * 32 + li) * GBK + k];
            a1[e] = Asm[(wm * 32 + 16 + li) * GBK + k];
            b0[e] = Bsm[(wn * 32 + li) * GBK + k];
            b1[e] = Bsm[(wn * 32 + 16 + li) * GBK + k];
        }
        acc[0][0] = __builtin_amdgcn_wmma_f32_16x16x32_bf16(false, a0, false, b0, (short)0, acc[0][0], false, false);
        acc[0][1] = __builtin_amdgcn_wmma_f32_16x16x32_bf16(false, a0, false, b1, (short)0, acc[0][1], false, false);
        acc[1][0] = __builtin_amdgcn_wmma_f32_16x16x32_bf16(false, a1, false, b0, (short)0, acc[1][0], false, false);
        acc[1][1] = __builtin_amdgcn_wmma_f32_16x16x32_bf16(false, a1, false, b1, (short)0, acc[1][1], false, false);
        __syncthreads();
    }

    // ---- epilogue: C/D layout — VGPR r holds row (r + 8g), col = li -------
    #pragma unroll
    for (int i = 0; i < 2; ++i) {
        #pragma unroll
        for (int j = 0; j < 2; ++j) {
            const int col = colBase + wn * 32 + j * 16 + li;
            #pragma unroll
            for (int r = 0; r < 8; ++r) {
                int row = rowBase + wm * 32 + i * 16 + g * 8 + r;
                float v = (col < N) ? (acc[i][j][r] + bias[col]) : 0.0f;
                if (mode == 0) {
                    if (col < N) outF[(size_t)row * N + col] = v;
                } else {
                    v = v > 0.f ? v : 0.f;
                    outB[(size_t)row * ldOutB + col] = (bf16)v;
                }
            }
        }
    }
}

// ---------------------------------------------------------------------------
// fp32 (Msrc x K) -> bf16 (rows x Kp), zero padded in both dims.
// Grid covers rows*Kp.
// ---------------------------------------------------------------------------
__global__ void conv_pad_bf16(const float* __restrict__ src, bf16* __restrict__ dst,
                              int Msrc, int K, int Kp)
{
    size_t idx = (size_t)blockIdx.x * blockDim.x + threadIdx.x;
    int r = (int)(idx / Kp), c = (int)(idx % Kp);
    dst[idx] = (r < Msrc && c < K) ? (bf16)src[(size_t)r * K + c] : (bf16)0.f;
}

// ---------------------------------------------------------------------------
// GRU recurrence scan (fp32, h resident in LDS). grid = 2 (dir), 640 threads.
// gi = precomputed x@Wih^T + bih, laid out (B*T, 3H) with row = b*T + t.
// ---------------------------------------------------------------------------
__global__ __launch_bounds__(640) void gru_scan(
    const float* __restrict__ giF, const float* __restrict__ giB,
    const float* __restrict__ WhhF, const float* __restrict__ WhhB,
    const float* __restrict__ bhhF, const float* __restrict__ bhhB,
    float* __restrict__ hcat)
{
    const int dir = blockIdx.x;
    const float* gi  = dir ? giB  : giF;
    const float* Whh = dir ? WhhB : WhhF;
    const float* bhh = dir ? bhhB : bhhF;

    __shared__ float h[B_SZ * NH];                // 25.6 KB
    for (int i = threadIdx.x; i < B_SZ * NH; i += 640) h[i] = 0.f;
    __syncthreads();

    float hn[10];
    for (int s = 0; s < T_SZ; ++s) {
        int t = dir ? (T_SZ - 1 - s) : s;
        int tn = dir ? (t - 1) : (t + 1);
        #pragma unroll 1
        for (int j = 0; j < 10; ++j) {
            int idx = threadIdx.x + j * 640;      // 6400 (b,k) pairs
            int b = idx / NH, k = idx % NH;
            const float* hb = &h[b * NH];
            float ghr = bhh[k], ghz = bhh[NH + k], ghn = bhh[2 * NH + k];
            const float* wr = &Whh[(size_t)k * NH];
            const float* wz = &Whh[(size_t)(NH + k) * NH];
            const float* wn = &Whh[(size_t)(2 * NH + k) * NH];
            for (int q = 0; q < NH; ++q) {
                float hv = hb[q];
                ghr += hv * wr[q];
                ghz += hv * wz[q];
                ghn += hv * wn[q];
            }
            const float* gr = &gi[((size_t)b * T_SZ + t) * (3 * NH)];
            if (tn >= 0 && tn < T_SZ)             // stream next step's gates
                __builtin_prefetch(&gi[((size_t)b * T_SZ + tn) * (3 * NH) + k], 0, 1);
            float r = 1.f / (1.f + __expf(-(gr[k] + ghr)));
            float z = 1.f / (1.f + __expf(-(gr[NH + k] + ghz)));
            float n = tanhf(gr[2 * NH + k] + r * ghn);
            hn[j] = (1.f - z) * n + z * hb[k];
        }
        __syncthreads();
        for (int j = 0; j < 10; ++j) h[threadIdx.x + j * 640] = hn[j];
        __syncthreads();
    }
    for (int i = threadIdx.x; i < B_SZ * NH; i += 640) {
        int b = i / NH, k = i % NH;
        hcat[b * (2 * NH) + dir * NH + k] = h[i];
    }
}

// ---------------------------------------------------------------------------
// encoder_m + gumbel-softmax. grid = B, 256 threads per row.
// ---------------------------------------------------------------------------
__global__ __launch_bounds__(256) void encoder_m(
    const float* __restrict__ hcat, const float* __restrict__ eps,
    const float* __restrict__ m1W, const float* __restrict__ m1b,
    const float* __restrict__ m2W, const float* __restrict__ m2b,
    const float* __restrict__ m3W, const float* __restrict__ m3b,
    float* __restrict__ alpha_out, float* __restrict__ y_out)
{
    __shared__ float hin[2 * NH];
    __shared__ float b1[1000];
    __shared__ float b2[500];
    __shared__ float sv[NS];
    const int b = blockIdx.x, tid = threadIdx.x;

    for (int i = tid; i < 2 * NH; i += 256) hin[i] = hcat[b * (2 * NH) + i];
    __syncthreads();
    for (int j = tid; j < 1000; j += 256) {
        float s = m1b[j]; const float* w = &m1W[(size_t)j * 200];
        for (int q = 0; q < 200; ++q) s += hin[q] * w[q];
        b1[j] = s > 0.f ? s : 0.f;
    }
    __syncthreads();
    for (int j = tid; j < 500; j += 256) {
        float s = m2b[j]; const float* w = &m2W[(size_t)j * 1000];
        for (int q = 0; q < 1000; ++q) s += b1[q] * w[q];
        b2[j] = s > 0.f ? s : 0.f;
    }
    __syncthreads();
    for (int j = tid; j < NS; j += 256) {
        float s = m3b[j]; const float* w = &m3W[(size_t)j * 500];
        for (int q = 0; q < 500; ++q) s += b2[q] * w[q];
        alpha_out[b * NS + j] = __expf(s);
        float e = eps[b * NS + j];
        float gum = -__logf(-__logf(e + 1e-20f) + 1e-20f);
        sv[j] = (gum + s) * 2.0f;                 // /TEMP, TEMP = 0.5
    }
    __syncthreads();
    if (tid == 0) {
        float mx = sv[0];
        for (int j = 1; j < NS; ++j) mx = fmaxf(mx, sv[j]);
        float sum = 0.f;
        for (int j = 0; j < NS; ++j) { float ev = __expf(sv[j] - mx); sv[j] = ev; sum += ev; }
        float inv = 1.f / sum;
        for (int j = 0; j < NS; ++j) y_out[b * NS + j] = sv[j] * inv;
    }
}

// ---------------------------------------------------------------------------
// encoder_Azb. grid = B, 256 threads per row.
// ---------------------------------------------------------------------------
__global__ __launch_bounds__(256) void encoder_azb(
    const float* __restrict__ y,
    const float* __restrict__ a1W, const float* __restrict__ a1b,
    const float* __restrict__ a2W, const float* __restrict__ a2b,
    const float* __restrict__ aAW, const float* __restrict__ aAb,
    const float* __restrict__ azW, const float* __restrict__ azb,
    const float* __restrict__ abW, const float* __restrict__ abb,
    float* __restrict__ Aout, float* __restrict__ dz_out,
    float* __restrict__ dz_ws, float* __restrict__ bv_ws)
{
    __shared__ float yv[NS];
    __shared__ float b1[1000];
    __shared__ float b2[1000];
    const int b = blockIdx.x, tid = threadIdx.x;

    for (int i = tid; i < NS; i += 256) yv[i] = y[b * NS + i];
    __syncthreads();
    for (int j = tid; j < 1000; j += 256) {
        float s = a1b[j]; const float* w = &a1W[(size_t)j * NS];
        for (int q = 0; q < NS; ++q) s += yv[q] * w[q];
        b1[j] = s > 0.f ? s : 0.f;
    }
    __syncthreads();
    for (int j = tid; j < 1000; j += 256) {
        float s = a2b[j]; const float* w = &a2W[(size_t)j * 1000];
        for (int q = 0; q < 1000; ++q) s += b1[q] * w[q];
        b2[j] = s > 0.f ? s : 0.f;
    }
    __syncthreads();
    if (tid < DETD * DETD) {
        float s = aAb[tid]; const float* w = &aAW[(size_t)tid * 1000];
        for (int q = 0; q < 1000; ++q) s += b2[q] * w[q];
        Aout[b * (DETD * DETD) + tid] = tanhf(s) * (1.0f / DETD);
    } else if (tid >= 32 && tid < 32 + DETD) {
        int j = tid - 32;
        float s = azb[j]; const float* w = &azW[(size_t)j * 1000];
        for (int q = 0; q < 1000; ++q) s += b2[q] * w[q];
        dz_out[b * DETD + j] = s;
        dz_ws[b * DETD + j] = s;
    } else if (tid >= 64 && tid < 64 + DETD) {
        int j = tid - 64;
        float s = abb[j]; const float* w = &abW[(size_t)j * 1000];
        for (int q = 0; q < 1000; ++q) s += b2[q] * w[q];
        bv_ws[b * DETD + j] = s;
    }
}

// ---------------------------------------------------------------------------
// Linear dynamics scan z_t = z_{t-1} @ A, write (z_t + b) as bf16 padded to 32.
// 1 block, 64 threads (one per batch).
// ---------------------------------------------------------------------------
__global__ __launch_bounds__(64) void lindyn(
    const float* __restrict__ Amat, const float* __restrict__ dz,
    const float* __restrict__ bv, bf16* __restrict__ zseqb)
{
    const int b = threadIdx.x;
    float A[DETD * DETD], z[DETD], bb[DETD];
    for (int i = 0; i < DETD * DETD; ++i) A[i] = Amat[b * DETD * DETD + i];
    for (int e = 0; e < DETD; ++e) { z[e] = dz[b * DETD + e]; bb[e] = bv[b * DETD + e]; }

    for (int t = 0; t < T_SZ; ++t) {
        if (t > 0) {
            float zn[DETD];
            #pragma unroll
            for (int e = 0; e < DETD; ++e) {
                float s = 0.f;
                #pragma unroll
                for (int d = 0; d < DETD; ++d) s += z[d] * A[d * DETD + e];
                zn[e] = s;
            }
            #pragma unroll
            for (int e = 0; e < DETD; ++e) z[e] = zn[e];
        }
        bf16* row = &zseqb[((size_t)b * T_SZ + t) * 32];
        #pragma unroll
        for (int e = 0; e < DETD; ++e) row[e] = (bf16)(z[e] + bb[e]);
        #pragma unroll
        for (int e = DETD; e < 32; ++e) row[e] = (bf16)0.f;
    }
}

// ---------------------------------------------------------------------------
// masked_fill(x==0 -> 0) on x_hat, and copy x to output slot 3
// ---------------------------------------------------------------------------
__global__ void mask_copy(const float* __restrict__ x, float* __restrict__ out0,
                          float* __restrict__ out3, size_t n)
{
    size_t i = (size_t)blockIdx.x * blockDim.x + threadIdx.x;
    if (i >= n) return;
    float xv = x[i];
    out3[i] = xv;
    if (xv == 0.f) out0[i] = 0.f;
}

// ---------------------------------------------------------------------------
// Host launch
// ---------------------------------------------------------------------------
extern "C" void kernel_launch(void* const* d_in, const int* in_sizes, int n_in,
                              void* d_out, int out_size, void* d_ws, size_t ws_size,
                              hipStream_t stream)
{
    (void)in_sizes; (void)n_in; (void)out_size; (void)ws_size;

    // ---- inputs (setup_inputs flatten order)
    const float* x      = (const float*)d_in[0];
    const float* eps    = (const float*)d_in[1];
    const float* Wih_f  = (const float*)d_in[2];
    const float* Whh_f  = (const float*)d_in[3];
    const float* bih_f  = (const float*)d_in[4];
    const float* bhh_f  = (const float*)d_in[5];
    const float* Wih_b  = (const float*)d_in[6];
    const float* Whh_b  = (const float*)d_in[7];
    const float* bih_b  = (const float*)d_in[8];
    const float* bhh_b  = (const float*)d_in[9];
    const float* m1W = (const float*)d_in[10]; const float* m1b = (const float*)d_in[11];
    const float* m2W = (const float*)d_in[12]; const float* m2b = (const float*)d_in[13];
    const float* m3W = (const float*)d_in[14]; const float* m3b = (const float*)d_in[15];
    const float* a1W = (const float*)d_in[16]; const float* a1b = (const float*)d_in[17];
    const float* a2W = (const float*)d_in[18]; const float* a2b = (const float*)d_in[19];
    const float* aAW = (const float*)d_in[20]; const float* aAb = (const float*)d_in[21];
    const float* azW = (const float*)d_in[22]; const float* azb = (const float*)d_in[23];
    const float* abW = (const float*)d_in[24]; const float* abb = (const float*)d_in[25];
    const float* d1W = (const float*)d_in[26]; const float* d1b = (const float*)d_in[27];
    const float* d2W = (const float*)d_in[28]; const float* d2b = (const float*)d_in[29];
    const float* d3W = (const float*)d_in[30]; const float* d3b = (const float*)d_in[31];

    // ---- outputs: (x_hat, alpha, dz, x) flattened
    float* out    = (float*)d_out;
    float* out0   = out;                                 // x_hat (B,T,129)
    float* oalpha = out + (size_t)BT * DDIM;             // (B,100)
    float* odz    = oalpha + (size_t)B_SZ * NS;          // (B,5)
    float* out3   = odz + (size_t)B_SZ * DETD;           // x copy

    // ---- workspace carve-up (256B aligned)
    char* base = (char*)d_ws;
    size_t off = 0;
    auto take = [&](size_t bytes) -> char* {
        char* p = base + off;
        off = (off + bytes + 255) & ~(size_t)255;
        return p;
    };
    bf16*  XB    = (bf16*) take((size_t)BT * 160 * 2);      // x, Kp=160
    bf16*  WIHF  = (bf16*) take((size_t)320 * 160 * 2);     // Np=320
    bf16*  WIHB  = (bf16*) take((size_t)320 * 160 * 2);
    float* GIF   = (float*)take((size_t)BT * 300 * 4);
    float* GIB   = (float*)take((size_t)BT * 300 * 4);
    float* HCAT  = (float*)take((size_t)B_SZ * 200 * 4);
    float* YWS   = (float*)take((size_t)B_SZ * NS * 4);
    float* AM    = (float*)take((size_t)B_SZ * 25 * 4);
    float* BV    = (float*)take((size_t)B_SZ * DETD * 4);
    float* DZW   = (float*)take((size_t)B_SZ * DETD * 4);
    bf16*  ZSEQ  = (bf16*) take((size_t)BT * 32 * 2);
    bf16*  D1WB  = (bf16*) take((size_t)512 * 32 * 2);      // Np=512
    bf16*  D2WB  = (bf16*) take((size_t)1024 * 512 * 2);    // Np=1024
    bf16*  D3WB  = (bf16*) take((size_t)192 * 1024 * 2);    // Np=192
    bf16*  H1B   = (bf16*) take((size_t)BT * 512 * 2);
    bf16*  H2B   = (bf16*) take((size_t)BT * 1024 * 2);

    const dim3 blk256(256);
    auto cgrid = [](size_t total) { return dim3((unsigned)((total + 255) / 256)); };

    // ---- stage 0: bf16 conversions (zero-padded rows and cols)
    conv_pad_bf16<<<cgrid((size_t)BT * 160),   blk256, 0, stream>>>(x,     XB,   BT,   DDIM, 160);
    conv_pad_bf16<<<cgrid((size_t)320 * 160),  blk256, 0, stream>>>(Wih_f, WIHF, 300,  DDIM, 160);
    conv_pad_bf16<<<cgrid((size_t)320 * 160),  blk256, 0, stream>>>(Wih_b, WIHB, 300,  DDIM, 160);
    conv_pad_bf16<<<cgrid((size_t)512 * 32),   blk256, 0, stream>>>(d1W,   D1WB, 500,  DETD, 32);
    conv_pad_bf16<<<cgrid((size_t)1024 * 512), blk256, 0, stream>>>(d2W,   D2WB, 1000, 500,  512);
    conv_pad_bf16<<<cgrid((size_t)192 * 1024), blk256, 0, stream>>>(d3W,   D3WB, DDIM, 1000, 1024);

    // ---- stage 1: GRU input projections (WMMA)  gi = x @ Wih^T + bih
    wmma_gemm<<<dim3(BT / GBM, 320 / GBN), blk256, 0, stream>>>(
        XB, WIHF, bih_f, GIF, nullptr, BT, 300, 320, 160, 160, 160, 0, 0);
    wmma_gemm<<<dim3(BT / GBM, 320 / GBN), blk256, 0, stream>>>(
        XB, WIHB, bih_b, GIB, nullptr, BT, 300, 320, 160, 160, 160, 0, 0);

    // ---- stage 2: GRU recurrence (both directions in one launch)
    gru_scan<<<dim3(2), dim3(640), 0, stream>>>(GIF, GIB, Whh_f, Whh_b, bhh_f, bhh_b, HCAT);

    // ---- stage 3: encoder_m + gumbel softmax
    encoder_m<<<dim3(B_SZ), blk256, 0, stream>>>(HCAT, eps, m1W, m1b, m2W, m2b, m3W, m3b,
                                                 oalpha, YWS);

    // ---- stage 4: encoder_Azb
    encoder_azb<<<dim3(B_SZ), blk256, 0, stream>>>(YWS, a1W, a1b, a2W, a2b,
                                                   aAW, aAb, azW, azb, abW, abb,
                                                   AM, odz, DZW, BV);

    // ---- stage 5: linear dynamics scan -> z_seq (bf16, padded K=32)
    lindyn<<<dim3(1), dim3(64), 0, stream>>>(AM, DZW, BV, ZSEQ);

    // ---- stage 6: decoder MLP (WMMA; bf16 activations, fp32 accumulate)
    wmma_gemm<<<dim3(BT / GBM, 512 / GBN), blk256, 0, stream>>>(
        ZSEQ, D1WB, d1b, nullptr, H1B, BT, 500, 512, 32, 32, 32, 512, 1);       // d1 + relu
    wmma_gemm<<<dim3(BT / GBM, 1024 / GBN), blk256, 0, stream>>>(
        H1B, D2WB, d2b, nullptr, H2B, BT, 1000, 1024, 512, 512, 512, 1024, 1);  // d2 + relu
    wmma_gemm<<<dim3(BT / GBM, 192 / GBN), blk256, 0, stream>>>(
        H2B, D3WB, d3b, out0, nullptr, BT, DDIM, 192, 1024, 1024, 1024, 0, 0);  // d3

    // ---- stage 7: mask x==0 and emit x copy
    mask_copy<<<cgrid((size_t)BT * DDIM), blk256, 0, stream>>>(x, out0, out3, (size_t)BT * DDIM);
}
